// MultiModelFusionSystem_3032246911413
// MI455X (gfx1250) — matrix operand order repaired and checked by reading, not compiled, not run on hardware
//
#include <hip/hip_runtime.h>
#include <hip/hip_bf16.h>

typedef __attribute__((ext_vector_type(16))) __bf16 v16bf;
typedef __attribute__((ext_vector_type(8)))  float  v8f;

#define E_    4
#define N_    8192
#define DIN_  1024
#define D_    512
#define L_    64
#define S_    128
#define MMEM_ 2
#define KMEM_ 32
#define H_    8
#define DH_   64
#define FF_   2048
#define HID_  1024

#define SD_INV 0.04419417382415922f   /* 1/sqrt(512) */

struct alignas(16) U4 { unsigned x, y, z, w; };

// low 32 bits of a flat pointer into LDS == workgroup-relative LDS byte offset
__device__ __forceinline__ unsigned lds_off_of(const void* p)
{
    return (unsigned)(unsigned long long)p;
}

// ---------------------------------------------------------------------------
// fp32 -> bf16 conversion
// ---------------------------------------------------------------------------
__global__ void cvt_bf16_kernel(const float* __restrict__ in, __bf16* __restrict__ out, size_t n)
{
    size_t i = (size_t)blockIdx.x * blockDim.x + threadIdx.x;
    if (i < n) out[i] = (__bf16)in[i];
}

// ---------------------------------------------------------------------------
// elementwise add: out = a + b
// ---------------------------------------------------------------------------
__global__ void add_kernel(const float* __restrict__ a, const float* __restrict__ b,
                           float* __restrict__ out, size_t n)
{
    size_t i = (size_t)blockIdx.x * blockDim.x + threadIdx.x;
    if (i < n) out[i] = a[i] + b[i];
}

// ---------------------------------------------------------------------------
// Batched bf16 WMMA GEMM:  C[e] = act(A[e] * B[e] + bias[e])
// A: [M,K] bf16 row-major, B: [K,N] bf16 row-major.
// Workgroup tile 128x64, BK=32; 8 waves in 4x2, each wave 32x32 (2x2 WMMA).
// A tiles: double-buffered LDS, filled by GLOBAL_LOAD_ASYNC_TO_LDS_B128 one
// K-step ahead (ASYNCcnt + barrier pipelining). B tile staged synchronously
// (transposed to [n][k]) with a prefetch of the next tile.
// ---------------------------------------------------------------------------
#define BM 128
#define BN 64
#define BK 32
#define LDAP 40   /* padded LDS stride (bf16 elems): 80B rows -> 16B aligned  */
#define LDBP 40

template<bool RELU, bool OUT_BF16, bool HAS_BIAS>
__global__ __launch_bounds__(256) void gemm_bf16_kernel(
    const __bf16* __restrict__ A, const __bf16* __restrict__ B,
    const float* __restrict__ bias,
    float* __restrict__ Cf, __bf16* __restrict__ Cb,
    int Mdim, int Ndim, int Kdim,
    long long sA, long long sB, long long sBias, long long sC)
{
    __shared__ __bf16 ldsA[2][BM * LDAP];
    __shared__ __bf16 ldsB[BN * LDBP];

    const int e = blockIdx.z;
    A += (long long)e * sA;
    B += (long long)e * sB;
    if (HAS_BIAS) bias += (long long)e * sBias;

    const int m0 = blockIdx.y * BM;
    const int n0 = blockIdx.x * BN;

    const int tid  = threadIdx.x;
    const int lane = tid & 31;
    const int wave = tid >> 5;
    const int wm   = wave >> 1;   // 0..3 -> M offset wm*32
    const int wn   = wave & 1;    // 0..1 -> N offset wn*32
    const int grp  = lane >> 4;   // lane group (0: lanes 0-15, 1: 16-31)
    const int ml   = lane & 15;

    // ---- per-thread A staging slots: 2 chunks of 16B (8 bf16) each ----
    const __bf16* aptr[2];   // global base (k0 added at issue time)
    unsigned      aoff[2][2];// [chunk][buf] LDS byte offsets
    #pragma unroll
    for (int c = 0; c < 2; ++c) {
        int slot = tid + c * 256;          // 0..511
        int row  = slot >> 2;              // 0..127
        int col  = (slot & 3) << 3;        // 0,8,16,24
        int gr   = m0 + row;
        if (gr >= Mdim) gr = Mdim - 1;     // clamp: rows >= Mdim never stored
        aptr[c]    = A + (size_t)gr * Kdim + col;
        aoff[c][0] = lds_off_of(&ldsA[0][row * LDAP + col]);
        aoff[c][1] = lds_off_of(&ldsA[1][row * LDAP + col]);
    }

    auto issueA = [&](int k0, int buf) {
        #pragma unroll
        for (int c = 0; c < 2; ++c) {
            unsigned long long ga = (unsigned long long)(aptr[c] + k0);
            asm volatile("global_load_async_to_lds_b128 %0, %1, off"
                         :: "v"(aoff[c][buf]), "v"(ga) : "memory");
        }
    };

    v8f acc[2][2] = {};

    const int nk = Kdim / BK;
    issueA(0, 0);                          // prologue: async-fill buffer 0

    for (int ik = 0; ik < nk; ++ik) {
        const int k0  = ik * BK;
        const int buf = ik & 1;
        __syncthreads();                   // all waves done reading ldsB / old bufs
        // ---- stage B tile (BK x BN) into LDS transposed: [n][k] ----
        {
            int row = tid >> 3;            // k: 0..31
            int col = (tid & 7) << 3;      // n: 0,8,...,56
            U4 val = *reinterpret_cast<const U4*>(B + (size_t)(k0 + row) * Ndim + n0 + col);
            union { U4 u; __bf16 h[8]; } cv;
            cv.u = val;
            #pragma unroll
            for (int j = 0; j < 8; ++j)
                ldsB[(col + j) * LDBP + row] = cv.h[j];
            if (ik + 1 < nk)
                __builtin_prefetch(B + (size_t)(k0 + BK + row) * Ndim + n0 + col, 0, 0);
        }
        asm volatile("s_wait_asynccnt 0x0" ::: "memory");  // my A loads for buf done
        __syncthreads();                   // everyone's A loads + B stores visible
        if (ik + 1 < nk)
            issueA(k0 + BK, buf ^ 1);      // async-fill next buffer while computing

        // ---- build fragments (ISA 7.12.2 16-bit A 16x32 / B 32x16 layouts) ----
        v16bf af[2], bfr[2];
        #pragma unroll
        for (int mt = 0; mt < 2; ++mt) {
            int row = wm * 32 + mt * 16 + ml;
            const __bf16* bse = &ldsA[buf][row * LDAP];
            #pragma unroll
            for (int v = 0; v < 8; ++v) {
                int kk = (v < 4) ? (2 * v + 8 * grp) : (16 + 2 * (v - 4) + 8 * grp);
                union { unsigned u; __bf16 h[2]; } cv;
                cv.u = *reinterpret_cast<const unsigned*>(bse + kk);
                af[mt][2 * v]     = cv.h[0];
                af[mt][2 * v + 1] = cv.h[1];
            }
        }
        #pragma unroll
        for (int nt = 0; nt < 2; ++nt) {
            int n = wn * 32 + nt * 16 + ml;
            const __bf16* bse = &ldsB[n * LDBP];
            #pragma unroll
            for (int v = 0; v < 8; ++v) {
                int kk = (v < 4) ? (2 * v + 8 * grp) : (16 + 2 * (v - 4) + 8 * grp);
                union { unsigned u; __bf16 h[2]; } cv;
                cv.u = *reinterpret_cast<const unsigned*>(bse + kk);
                bfr[nt][2 * v]     = cv.h[0];
                bfr[nt][2 * v + 1] = cv.h[1];
            }
        }
        #pragma unroll
        for (int mt = 0; mt < 2; ++mt)
            #pragma unroll
            for (int nt = 0; nt < 2; ++nt)
                acc[mt][nt] = __builtin_amdgcn_wmma_f32_16x16x32_bf16(
                    false, af[mt], false, bfr[nt], (short)0, acc[mt][nt], false, false);
    }

    if (OUT_BF16) Cb += (long long)e * sC;
    else          Cf += (long long)e * sC;

    // ---- epilogue (C/D layout: lane -> n=ml, VGPR r -> m = r + 8*grp) ----
    #pragma unroll
    for (int mt = 0; mt < 2; ++mt) {
        #pragma unroll
        for (int nt = 0; nt < 2; ++nt) {
            int col = n0 + wn * 32 + nt * 16 + ml;
            float bv = 0.f;
            if (HAS_BIAS) bv = bias[col];
            #pragma unroll
            for (int r = 0; r < 8; ++r) {
                int row = m0 + wm * 32 + mt * 16 + r + 8 * grp;
                if (row < Mdim) {
                    float vv = acc[mt][nt][r] + bv;
                    if (RELU) vv = fmaxf(vv, 0.f);
                    if (OUT_BF16) Cb[(size_t)row * Ndim + col] = (__bf16)vv;
                    else          Cf[(size_t)row * Ndim + col] = vv;
                }
            }
        }
    }
}

// ---------------------------------------------------------------------------
// LayerNorm over last dim (Ddim<=512), one block per row.
// out = ln(X + R) * g + b  (R/g/b/outF/outB optional via null).
// ---------------------------------------------------------------------------
__global__ __launch_bounds__(256) void layernorm_kernel(
    const float* __restrict__ X, const float* __restrict__ R,
    const float* __restrict__ g, const float* __restrict__ b,
    float* __restrict__ outF, __bf16* __restrict__ outB,
    int rowsPerExpert, int Ddim)
{
    __shared__ float red[256];
    const int row = blockIdx.x;
    const int e   = row / rowsPerExpert;
    const int tid = threadIdx.x;
    const float* xr = X + (size_t)row * Ddim;
    const float* rr = R ? R + (size_t)row * Ddim : nullptr;

    float xs[2];
    int cnt = 0;
    float sum = 0.f;
    for (int d = tid; d < Ddim; d += 256) {
        float v = xr[d] + (rr ? rr[d] : 0.f);
        xs[cnt++] = v;
        sum += v;
    }
    red[tid] = sum; __syncthreads();
    for (int off = 128; off > 0; off >>= 1) { if (tid < off) red[tid] += red[tid + off]; __syncthreads(); }
    float mean = red[0] / (float)Ddim;
    __syncthreads();

    float s2 = 0.f;
    for (int i = 0; i < cnt; ++i) { float d = xs[i] - mean; s2 += d * d; }
    red[tid] = s2; __syncthreads();
    for (int off = 128; off > 0; off >>= 1) { if (tid < off) red[tid] += red[tid + off]; __syncthreads(); }
    float inv = rsqrtf(red[0] / (float)Ddim + 1e-5f);

    cnt = 0;
    for (int d = tid; d < Ddim; d += 256) {
        float v = (xs[cnt++] - mean) * inv;
        if (g) v = v * g[(size_t)e * Ddim + d] + b[(size_t)e * Ddim + d];
        if (outF) outF[(size_t)row * Ddim + d] = v;
        if (outB) outB[(size_t)row * Ddim + d] = (__bf16)v;
    }
}

// ---------------------------------------------------------------------------
// LSTCWA segment cross-attention: one block per (e,l).
// zo[e,l,:] = softmax(clip(q.k/sd)) @ v   over S=128 segments, D=512.
// ---------------------------------------------------------------------------
__global__ __launch_bounds__(256) void lstcwa_attn_kernel(
    const float* __restrict__ q, const __bf16* __restrict__ k,
    const __bf16* __restrict__ v, float* __restrict__ zo)
{
    __shared__ float sc[S_];
    __shared__ float red[256];
    const int e = blockIdx.x >> 6, l = blockIdx.x & 63;
    const int tid = threadIdx.x;
    const float*  qr = q + ((size_t)e * L_ + l) * D_;
    const __bf16* kb = k + ((size_t)e * L_ + l) * S_ * D_;
    const __bf16* vb = v + ((size_t)e * L_ + l) * S_ * D_;

    const int seg = tid >> 1, half = tid & 1;    // 2 threads per segment
    {
        const __bf16* kr = kb + (size_t)seg * D_ + half * (D_ / 2);
        const float*  qh = qr + half * (D_ / 2);
        float p = 0.f;
        for (int i = 0; i < D_ / 2; ++i) p += qh[i] * (float)kr[i];
        red[tid] = p;
    }
    __syncthreads();
    if (half == 0) {
        float s = (red[tid] + red[tid + 1]) * SD_INV;
        sc[seg] = fminf(fmaxf(s, -5.f), 5.f);
    }
    __syncthreads();
    if (tid == 0) {
        float mx = -1e30f;
        for (int j = 0; j < S_; ++j) mx = fmaxf(mx, sc[j]);
        float sm = 0.f;
        for (int j = 0; j < S_; ++j) { float ex = __expf(sc[j] - mx); sc[j] = ex; sm += ex; }
        float inv = 1.f / sm;
        for (int j = 0; j < S_; ++j) sc[j] *= inv;
    }
    __syncthreads();
    for (int d = tid; d < D_; d += 256) {
        float acc = 0.f;
        for (int j = 0; j < S_; ++j) acc += sc[j] * (float)vb[(size_t)j * D_ + d];
        zo[((size_t)e * L_ + l) * D_ + d] = acc;
    }
}

// ---------------------------------------------------------------------------
// Context cross-attention: xin = tn + softmax(clip(tn.ctx/sd)) @ ctx
// ctx = tn[other expert]; one block per (e,l).
// ---------------------------------------------------------------------------
__global__ __launch_bounds__(256) void cross_attn_kernel(
    const float* __restrict__ tn, float* __restrict__ xin)
{
    __shared__ float sc[L_];
    __shared__ float red[256];
    const int e = blockIdx.x >> 6, l = blockIdx.x & 63;
    const int ce = (e == 0) ? 1 : 0;
    const int tid = threadIdx.x;
    const float* trow = tn + ((size_t)e * L_ + l) * D_;
    const float* ctx  = tn + (size_t)ce * L_ * D_;

    const int m = tid >> 2, part = tid & 3;      // 4 threads per ctx row
    {
        const float* cr = ctx + (size_t)m * D_ + part * 128;
        const float* qr = trow + part * 128;
        float p = 0.f;
        for (int i = 0; i < 128; ++i) p += qr[i] * cr[i];
        red[tid] = p;
    }
    __syncthreads();
    if (part == 0) {
        float s = (red[tid] + red[tid + 1] + red[tid + 2] + red[tid + 3]) * SD_INV;
        sc[m] = fminf(fmaxf(s, -5.f), 5.f);
    }
    __syncthreads();
    if (tid == 0) {
        float mx = -1e30f;
        for (int j = 0; j < L_; ++j) mx = fmaxf(mx, sc[j]);
        float sm = 0.f;
        for (int j = 0; j < L_; ++j) { float ex = __expf(sc[j] - mx); sc[j] = ex; sm += ex; }
        float inv = 1.f / sm;
        for (int j = 0; j < L_; ++j) sc[j] *= inv;
    }
    __syncthreads();
    for (int d = tid; d < D_; d += 256) {
        float acc = trow[d];
        for (int j = 0; j < L_; ++j) acc += sc[j] * ctx[(size_t)j * D_ + d];
        xin[((size_t)e * L_ + l) * D_ + d] = acc;
    }
}

// ---------------------------------------------------------------------------
// Multi-head self-attention (64x64, DH=64, no clip): one block per (e,h).
// qkv layout: [E, L, 3*D] with q|k|v, head h at column offset h*64.
// ---------------------------------------------------------------------------
__global__ __launch_bounds__(256) void mha_kernel(
    const float* __restrict__ qkv, float* __restrict__ ao)
{
    __shared__ float ks[L_ * DH_];
    __shared__ float vs[L_ * DH_];
    __shared__ float sc[L_ * L_];
    const int e = blockIdx.x >> 3, h = blockIdx.x & 7;
    const int tid = threadIdx.x;
    const float* bse = qkv + (size_t)e * L_ * (3 * D_);

    for (int i = tid; i < L_ * DH_; i += 256) {
        int l = i >> 6, d = i & 63;
        ks[i] = bse[(size_t)l * (3 * D_) + D_     + h * DH_ + d];
        vs[i] = bse[(size_t)l * (3 * D_) + 2 * D_ + h * DH_ + d];
    }
    __syncthreads();
    for (int i = tid; i < L_ * L_; i += 256) {
        int l = i >> 6, m = i & 63;
        const float* qrow = bse + (size_t)l * (3 * D_) + h * DH_;
        float s = 0.f;
        for (int d = 0; d < DH_; ++d) s += qrow[d] * ks[m * DH_ + d];
        sc[i] = s * 0.125f;   // 1/sqrt(64)
    }
    __syncthreads();
    if (tid < L_) {
        float mx = -1e30f;
        for (int m = 0; m < L_; ++m) mx = fmaxf(mx, sc[tid * L_ + m]);
        float sm = 0.f;
        for (int m = 0; m < L_; ++m) { float ex = __expf(sc[tid * L_ + m] - mx); sc[tid * L_ + m] = ex; sm += ex; }
        float inv = 1.f / sm;
        for (int m = 0; m < L_; ++m) sc[tid * L_ + m] *= inv;
    }
    __syncthreads();
    for (int i = tid; i < L_ * DH_; i += 256) {
        int l = i >> 6, d = i & 63;
        float acc = 0.f;
        for (int m = 0; m < L_; ++m) acc += sc[l * L_ + m] * vs[m * DH_ + d];
        ao[((size_t)e * L_ + l) * D_ + h * DH_ + d] = acc;
    }
}

// ---------------------------------------------------------------------------
// Hierarchical memory retrieval: one block per (e,l).
// retr[e,l, mm*512+d] = softmax_k(clip(eo.mem/sd)) @ mem   (bf16 out for GEMM)
// ---------------------------------------------------------------------------
__global__ __launch_bounds__(256) void mem_retrieve_kernel(
    const float* __restrict__ eo, const float* __restrict__ mem,
    __bf16* __restrict__ retr)
{
    __shared__ float sc[KMEM_];
    __shared__ float red[256];
    const int e = blockIdx.x >> 6, l = blockIdx.x & 63;
    const int tid = threadIdx.x;
    const float* er = eo + ((size_t)e * L_ + l) * D_;

    for (int mm = 0; mm < MMEM_; ++mm) {
        const float* mb = mem + (size_t)mm * KMEM_ * D_;
        const int kk = tid >> 3, part = tid & 7;     // 8 threads per memory slot
        {
            const float* mr = mb + (size_t)kk * D_ + part * 64;
            const float* qr = er + part * 64;
            float p = 0.f;
            for (int i = 0; i < 64; ++i) p += qr[i] * mr[i];
            red[tid] = p;
        }
        __syncthreads();
        if (part == 0) {
            float s = 0.f;
            for (int j = 0; j < 8; ++j) s += red[tid + j];
            s *= SD_INV;
            sc[kk] = fminf(fmaxf(s, -5.f), 5.f);
        }
        __syncthreads();
        if (tid == 0) {
            float mx = -1e30f;
            for (int j = 0; j < KMEM_; ++j) mx = fmaxf(mx, sc[j]);
            float sm = 0.f;
            for (int j = 0; j < KMEM_; ++j) { float ex = __expf(sc[j] - mx); sc[j] = ex; sm += ex; }
            float inv = 1.f / sm;
            for (int j = 0; j < KMEM_; ++j) sc[j] *= inv;
        }
        __syncthreads();
        for (int d = tid; d < D_; d += 256) {
            float acc = 0.f;
            for (int j = 0; j < KMEM_; ++j) acc += sc[j] * mb[(size_t)j * D_ + d];
            retr[((size_t)e * L_ + l) * (MMEM_ * D_) + mm * D_ + d] = (__bf16)acc;
        }
        __syncthreads();
    }
}

// ---------------------------------------------------------------------------
// Mean-pool over L, gate logits = clip(pooled @ gg), gw = softmax over E.
// Single block.
// ---------------------------------------------------------------------------
__global__ __launch_bounds__(256) void pool_gate_kernel(
    const float* __restrict__ eout, const float* __restrict__ gg,
    float* __restrict__ gw)
{
    __shared__ float logits[E_];
    __shared__ float red[256];
    const int tid = threadIdx.x;
    for (int e = 0; e < E_; ++e) {
        float part = 0.f;
        for (int d = tid; d < D_; d += 256) {
            float s = 0.f;
            for (int l = 0; l < L_; ++l) s += eout[((size_t)e * L_ + l) * D_ + d];
            part += (s * (1.f / (float)L_)) * gg[d];
        }
        red[tid] = part; __syncthreads();
        for (int off = 128; off > 0; off >>= 1) { if (tid < off) red[tid] += red[tid + off]; __syncthreads(); }
        if (tid == 0) logits[e] = fminf(fmaxf(red[0], -5.f), 5.f);
        __syncthreads();
    }
    if (tid == 0) {
        float mx = logits[0];
        for (int e = 1; e < E_; ++e) mx = fmaxf(mx, logits[e]);
        float ex[E_]; float sm = 0.f;
        for (int e = 0; e < E_; ++e) { ex[e] = __expf(logits[e] - mx); sm += ex[e]; }
        float inv = 1.f / sm;
        for (int e = 0; e < E_; ++e) gw[e] = ex[e] * inv;
    }
}

// ---------------------------------------------------------------------------
// fused[l,d] = sum_e gw[e] * eout[e,l,d]
// ---------------------------------------------------------------------------
__global__ void fuse_kernel(const float* __restrict__ eout, const float* __restrict__ gw,
                            float* __restrict__ out)
{
    int i = blockIdx.x * blockDim.x + threadIdx.x;
    if (i < L_ * D_) {
        float acc = 0.f;
        for (int e = 0; e < E_; ++e) acc += gw[e] * eout[(size_t)e * L_ * D_ + i];
        out[i] = acc;
    }
}

// ---------------------------------------------------------------------------
// Host orchestration
// ---------------------------------------------------------------------------
extern "C" void kernel_launch(void* const* d_in, const int* in_sizes, int n_in,
                              void* d_out, int out_size, void* d_ws, size_t ws_size,
                              hipStream_t stream)
{
    (void)in_sizes; (void)n_in; (void)out_size; (void)ws_size;

    const float* x          = (const float*)d_in[0];
    const float* al_w1      = (const float*)d_in[1];
    const float* al_b1      = (const float*)d_in[2];
    const float* al_w2      = (const float*)d_in[3];
    const float* al_b2      = (const float*)d_in[4];
    const float* al_g       = (const float*)d_in[5];
    const float* al_bt      = (const float*)d_in[6];
    const float* z          = (const float*)d_in[7];
    const float* wq         = (const float*)d_in[8];
    const float* wk         = (const float*)d_in[9];
    const float* wv         = (const float*)d_in[10];
    const float* wo         = (const float*)d_in[11];
    const float* bo         = (const float*)d_in[12];
    const float* attn_in_w  = (const float*)d_in[13];
    const float* attn_in_b  = (const float*)d_in[14];
    const float* attn_out_w = (const float*)d_in[15];
    const float* attn_out_b = (const float*)d_in[16];
    const float* ln1_g      = (const float*)d_in[17];
    const float* ln1_b      = (const float*)d_in[18];
    const float* ff_w1      = (const float*)d_in[19];
    const float* ff_b1      = (const float*)d_in[20];
    const float* ff_w2      = (const float*)d_in[21];
    const float* ff_b2      = (const float*)d_in[22];
    const float* ln2_g      = (const float*)d_in[23];
    const float* ln2_b      = (const float*)d_in[24];
    const float* memories   = (const float*)d_in[25];
    const float* mem_agg_w  = (const float*)d_in[26];
    const float* mem_agg_b  = (const float*)d_in[27];
    const float* gg         = (const float*)d_in[28];
    float* out = (float*)d_out;

    char* wsbase = (char*)d_ws;
    size_t off = 0;
    auto alloc = [&](size_t bytes) -> char* {
        char* r = wsbase + off;
        off += (bytes + 255) & ~(size_t)255;
        return r;
    };

    // big regions (with reuse)
    __bf16* xb   = (__bf16*)alloc((size_t)E_ * N_ * DIN_ * 2);   // -> reused as kb|vb
    __bf16* w1b  = (__bf16*)alloc((size_t)E_ * DIN_ * HID_ * 2);
    __bf16* hb   = (__bf16*)alloc((size_t)E_ * N_ * HID_ * 2);   // -> reused as fb
    __bf16* w2b  = (__bf16*)alloc((size_t)E_ * HID_ * D_ * 2);
    float*  yf   = (float*) alloc((size_t)E_ * N_ * D_ * 4);
    // bf16 weights
    __bf16* wqb  = (__bf16*)alloc((size_t)E_ * D_ * D_ * 2);
    __bf16* wkb  = (__bf16*)alloc((size_t)E_ * D_ * D_ * 2);
    __bf16* wvb  = (__bf16*)alloc((size_t)E_ * D_ * D_ * 2);
    __bf16* wob  = (__bf16*)alloc((size_t)E_ * D_ * D_ * 2);
    __bf16* aiwb = (__bf16*)alloc((size_t)E_ * D_ * 3 * D_ * 2);
    __bf16* aowb = (__bf16*)alloc((size_t)E_ * D_ * D_ * 2);
    __bf16* f1wb = (__bf16*)alloc((size_t)E_ * D_ * FF_ * 2);
    __bf16* f2wb = (__bf16*)alloc((size_t)E_ * FF_ * D_ * 2);
    __bf16* magb = (__bf16*)alloc((size_t)MMEM_ * D_ * D_ * 2);
    // small activations
    __bf16* zbb  = (__bf16*)alloc((size_t)E_ * L_ * D_ * 2);
    float*  qf   = (float*) alloc((size_t)E_ * L_ * D_ * 4);
    float*  zof  = (float*) alloc((size_t)E_ * L_ * D_ * 4);
    __bf16* zob  = (__bf16*)alloc((size_t)E_ * L_ * D_ * 2);
    float*  tokf = (float*) alloc((size_t)E_ * L_ * D_ * 4);
    float*  tnf  = (float*) alloc((size_t)E_ * L_ * D_ * 4);
    float*  xinf = (float*) alloc((size_t)E_ * L_ * D_ * 4);
    __bf16* xinb = (__bf16*)alloc((size_t)E_ * L_ * D_ * 2);
    float*  qkvf = (float*) alloc((size_t)E_ * L_ * 3 * D_ * 4);
    float*  aof  = (float*) alloc((size_t)E_ * L_ * D_ * 4);
    __bf16* aob  = (__bf16*)alloc((size_t)E_ * L_ * D_ * 2);
    float*  prjf = (float*) alloc((size_t)E_ * L_ * D_ * 4);
    float*  x1f  = (float*) alloc((size_t)E_ * L_ * D_ * 4);
    __bf16* x1b  = (__bf16*)alloc((size_t)E_ * L_ * D_ * 2);
    __bf16* ffhb = (__bf16*)alloc((size_t)E_ * L_ * FF_ * 2);
    float*  ffof = (float*) alloc((size_t)E_ * L_ * D_ * 4);
    float*  eofb = (float*) alloc((size_t)E_ * L_ * D_ * 4);
    __bf16* retb = (__bf16*)alloc((size_t)E_ * L_ * MMEM_ * D_ * 2);
    float*  ragf = (float*) alloc((size_t)E_ * L_ * D_ * 4);
    float*  eouf = (float*) alloc((size_t)E_ * L_ * D_ * 4);

    // region reuse (lifetimes disjoint):
    __bf16* kb = xb;                                // after x consumed by gemm1
    __bf16* vb = xb + (size_t)E_ * N_ * D_;
    __bf16* fb = hb;                                // after h consumed by gemm2

    auto cvt = [&](const float* src, __bf16* dst, size_t n) {
        cvt_bf16_kernel<<<dim3((unsigned)((n + 255) / 256)), 256, 0, stream>>>(src, dst, n);
    };

    // ---- precision conversion ----
    cvt(x, xb, (size_t)E_ * N_ * DIN_);
    cvt(al_w1, w1b, (size_t)E_ * DIN_ * HID_);
    cvt(al_w2, w2b, (size_t)E_ * HID_ * D_);
    cvt(wq, wqb, (size_t)E_ * D_ * D_);
    cvt(wk, wkb, (size_t)E_ * D_ * D_);
    cvt(wv, wvb, (size_t)E_ * D_ * D_);
    cvt(wo, wob, (size_t)E_ * D_ * D_);
    cvt(attn_in_w, aiwb, (size_t)E_ * D_ * 3 * D_);
    cvt(attn_out_w, aowb, (size_t)E_ * D_ * D_);
    cvt(ff_w1, f1wb, (size_t)E_ * D_ * FF_);
    cvt(ff_w2, f2wb, (size_t)E_ * FF_ * D_);
    cvt(mem_agg_w, magb, (size_t)MMEM_ * D_ * D_);
    cvt(z, zbb, (size_t)E_ * L_ * D_);

    // ---- MultiDimAligner ----
    gemm_bf16_kernel<true, true, true><<<dim3(HID_ / BN, N_ / BM, E_), 256, 0, stream>>>(
        xb, w1b, al_b1, nullptr, hb, N_, HID_, DIN_,
        (long long)N_ * DIN_, (long long)DIN_ * HID_, HID_, (long long)N_ * HID_);
    gemm_bf16_kernel<false, false, true><<<dim3(D_ / BN, N_ / BM, E_), 256, 0, stream>>>(
        hb, w2b, al_b2, yf, nullptr, N_, D_, HID_,
        (long long)N_ * HID_, (long long)HID_ * D_, D_, (long long)N_ * D_);
    layernorm_kernel<<<E_ * N_, 256, 0, stream>>>(yf, nullptr, al_g, al_bt, yf, nullptr, N_, D_);
    layernorm_kernel<<<E_ * N_, 256, 0, stream>>>(yf, nullptr, nullptr, nullptr, nullptr, fb, N_, D_);

    // ---- LSTCWA ----
    gemm_bf16_kernel<false, true, false><<<dim3(D_ / BN, N_ / BM, E_), 256, 0, stream>>>(
        fb, wkb, nullptr, nullptr, kb, N_, D_, D_,
        (long long)N_ * D_, (long long)D_ * D_, 0, (long long)N_ * D_);
    gemm_bf16_kernel<false, true, false><<<dim3(D_ / BN, N_ / BM, E_), 256, 0, stream>>>(
        fb, wvb, nullptr, nullptr, vb, N_, D_, D_,
        (long long)N_ * D_, (long long)D_ * D_, 0, (long long)N_ * D_);
    gemm_bf16_kernel<false, false, false><<<dim3(D_ / BN, 1, E_), 256, 0, stream>>>(
        zbb, wqb, nullptr, qf, nullptr, L_, D_, D_,
        (long long)L_ * D_, (long long)D_ * D_, 0, (long long)L_ * D_);
    lstcwa_attn_kernel<<<E_ * L_, 256, 0, stream>>>(qf, kb, vb, zof);
    cvt(zof, zob, (size_t)E_ * L_ * D_);
    gemm_bf16_kernel<false, false, true><<<dim3(D_ / BN, 1, E_), 256, 0, stream>>>(
        zob, wob, bo, tokf, nullptr, L_, D_, D_,
        (long long)L_ * D_, (long long)D_ * D_, D_, (long long)L_ * D_);

    // ---- HierarchicalMemoryMoE: context cross-attention ----
    layernorm_kernel<<<E_ * L_, 256, 0, stream>>>(tokf, nullptr, nullptr, nullptr, tnf, nullptr, L_, D_);
    cross_attn_kernel<<<E_ * L_, 256, 0, stream>>>(tnf, xinf);
    cvt(xinf, xinb, (size_t)E_ * L_ * D_);

    // ---- TransformerEncoderLayer (post-norm) ----
    gemm_bf16_kernel<false, false, true><<<dim3(3 * D_ / BN, 1, E_), 256, 0, stream>>>(
        xinb, aiwb, attn_in_b, qkvf, nullptr, L_, 3 * D_, D_,
        (long long)L_ * D_, (long long)D_ * 3 * D_, 3 * D_, (long long)L_ * 3 * D_);
    mha_kernel<<<E_ * H_, 256, 0, stream>>>(qkvf, aof);
    cvt(aof, aob, (size_t)E_ * L_ * D_);
    gemm_bf16_kernel<false, false, true><<<dim3(D_ / BN, 1, E_), 256, 0, stream>>>(
        aob, aowb, attn_out_b, prjf, nullptr, L_, D_, D_,
        (long long)L_ * D_, (long long)D_ * D_, D_, (long long)L_ * D_);
    layernorm_kernel<<<E_ * L_, 256, 0, stream>>>(prjf, xinf, ln1_g, ln1_b, x1f, x1b, L_, D_);
    gemm_bf16_kernel<true, true, true><<<dim3(FF_ / BN, 1, E_), 256, 0, stream>>>(
        x1b, f1wb, ff_b1, nullptr, ffhb, L_, FF_, D_,
        (long long)L_ * D_, (long long)D_ * FF_, FF_, (long long)L_ * FF_);
    gemm_bf16_kernel<false, false, true><<<dim3(D_ / BN, 1, E_), 256, 0, stream>>>(
        ffhb, f2wb, ff_b2, ffof, nullptr, L_, D_, FF_,
        (long long)L_ * FF_, (long long)FF_ * D_, D_, (long long)L_ * D_);
    layernorm_kernel<<<E_ * L_, 256, 0, stream>>>(ffof, x1f, ln2_g, ln2_b, eofb, nullptr, L_, D_);

    // ---- memory retrieval + aggregation ----
    mem_retrieve_kernel<<<E_ * L_, 256, 0, stream>>>(eofb, memories, retb);
    gemm_bf16_kernel<false, false, true><<<dim3(D_ / BN, (E_ * L_ + BM - 1) / BM, 1), 256, 0, stream>>>(
        retb, magb, mem_agg_b, ragf, nullptr, E_ * L_, D_, MMEM_ * D_, 0, 0, 0, 0);
    add_kernel<<<(E_ * L_ * D_ + 255) / 256, 256, 0, stream>>>(ragf, eofb, eouf, (size_t)E_ * L_ * D_);

    // ---- gating fusion ----
    pool_gate_kernel<<<1, 256, 0, stream>>>(eouf, gg, out + (size_t)L_ * D_);
    fuse_kernel<<<(L_ * D_ + 255) / 256, 256, 0, stream>>>(eouf, out + (size_t)L_ * D_, out);
}
